// SummarizationModel_25804163514472
// MI455X (gfx1250) — matrix operand order repaired and checked by reading, not compiled, not run on hardware
//
#include <hip/hip_runtime.h>
#include <hip/hip_bf16.h>

// ---------------------------------------------------------------------------
// R-GCN (3 layers, basis decomposition) for MI455X / gfx1250 (wave32, WMMA).
// Edge transform batched 16-wide per relation via v_wmma_f32_16x16x32_f16;
// weights held register-resident across 4 tiles (64 edges) per wave.
// ---------------------------------------------------------------------------

typedef __attribute__((ext_vector_type(16))) _Float16 v16h;
typedef __attribute__((ext_vector_type(8)))  _Float16 v8h;
typedef __attribute__((ext_vector_type(8)))  float    v8f;

#define N_NODES 50000
#define N_EDGES 500000
#define H_DIM   64
#define OUT_DIM 16
#define N_RELS  64
#define N_BASES 8

// ---------------------------------------------------------------------------
// Weight prep: Wt[r][o][i] = sum_b C[r,b] * B[b][i][o]   (f16, transposed so a
// WMMA B-fragment (16 contiguous K values for one output column) is one
// contiguous 32-byte load).
// ---------------------------------------------------------------------------
__global__ void compute_W_kernel(const float* __restrict__ C1, const float* __restrict__ B1,
                                 const float* __restrict__ C2, const float* __restrict__ B2,
                                 _Float16* __restrict__ W1t, _Float16* __restrict__ W2t) {
    int r = blockIdx.x;
    float c1[N_BASES], c2[N_BASES];
#pragma unroll
    for (int b = 0; b < N_BASES; ++b) { c1[b] = C1[r * N_BASES + b]; c2[b] = C2[r * N_BASES + b]; }
    for (int idx = threadIdx.x; idx < H_DIM * H_DIM; idx += blockDim.x) {
        int o = idx >> 6, i = idx & 63;
        float acc = 0.f;
#pragma unroll
        for (int b = 0; b < N_BASES; ++b) acc += c1[b] * B1[(b * H_DIM + i) * H_DIM + o];
        W1t[((size_t)r * H_DIM + o) * H_DIM + i] = (_Float16)acc;
    }
    for (int idx = threadIdx.x; idx < OUT_DIM * H_DIM; idx += blockDim.x) {
        int o = idx >> 6, i = idx & 63;
        float acc = 0.f;
#pragma unroll
        for (int b = 0; b < N_BASES; ++b) acc += c2[b] * B2[(b * H_DIM + i) * OUT_DIM + o];
        W2t[((size_t)r * OUT_DIM + o) * H_DIM + i] = (_Float16)acc;
    }
}

__global__ void prep_L_kernel(const float* __restrict__ L1, const float* __restrict__ L2,
                              _Float16* __restrict__ L1t, _Float16* __restrict__ L2t) {
    for (int idx = threadIdx.x; idx < H_DIM * H_DIM; idx += blockDim.x) {
        int o = idx >> 6, i = idx & 63;
        L1t[o * H_DIM + i] = (_Float16)L1[i * H_DIM + o];
    }
    for (int idx = threadIdx.x; idx < OUT_DIM * H_DIM; idx += blockDim.x) {
        int o = idx >> 6, i = idx & 63;
        L2t[o * H_DIM + i] = (_Float16)L2[i * OUT_DIM + o];
    }
}

// ---------------------------------------------------------------------------
// Edge binning by relation: histogram -> scan -> scatter.
// ---------------------------------------------------------------------------
__global__ void zero64_kernel(int* __restrict__ counts, int* __restrict__ cursor) {
    int i = threadIdx.x;
    if (i < N_RELS) { counts[i] = 0; cursor[i] = 0; }
}

__global__ void hist_kernel(const int* __restrict__ etype, int* __restrict__ counts) {
    int e = blockIdx.x * blockDim.x + threadIdx.x;
    if (e < N_EDGES) atomicAdd(&counts[etype[e]], 1);
}

// binOff: edge offsets; qtileOff: offsets in units of 64-edge quad-tiles
__global__ void scan_kernel(const int* __restrict__ counts,
                            int* __restrict__ binOff, int* __restrict__ qtileOff) {
    if (threadIdx.x == 0 && blockIdx.x == 0) {
        int b = 0, q = 0;
        binOff[0] = 0; qtileOff[0] = 0;
        for (int r = 0; r < N_RELS; ++r) {
            b += counts[r];
            q += (counts[r] + 63) >> 6;
            binOff[r + 1] = b;
            qtileOff[r + 1] = q;
        }
    }
}

__global__ void scatter_kernel(const int* __restrict__ etype, const int* __restrict__ binOff,
                               int* __restrict__ cursor, int* __restrict__ edgeIds) {
    int e = blockIdx.x * blockDim.x + threadIdx.x;
    if (e >= N_EDGES) return;
    int r = etype[e];
    int pos = atomicAdd(&cursor[r], 1);
    edgeIds[binOff[r] + pos] = e;
}

// ---------------------------------------------------------------------------
// Layer 0: h0 = L0 + b0 ; += scatter( norm * sum_b C0[r,b]*B0[b,src,:] )
// (feats == arange, so the embedding lookup is just the src node id)
// ---------------------------------------------------------------------------
__global__ void init_h0_kernel(const float* __restrict__ L0, const float* __restrict__ b0,
                               float* __restrict__ h0) {
    int i = blockIdx.x * blockDim.x + threadIdx.x;
    if (i < N_NODES * H_DIM) h0[i] = L0[i] + b0[i & 63];
}

__global__ void edge_layer0_kernel(const int* __restrict__ src, const int* __restrict__ dst,
                                   const int* __restrict__ etype, const float* __restrict__ norm,
                                   const float* __restrict__ B0, const float* __restrict__ C0,
                                   float* __restrict__ h0) {
    long long t = (long long)blockIdx.x * blockDim.x + threadIdx.x;
    int e = (int)(t >> 6);
    int dim = (int)(t & 63);
    if (e >= N_EDGES) return;
    int s = src[e], d = dst[e], r = etype[e];
    float nrm = norm[e];
    float acc = 0.f;
#pragma unroll
    for (int b = 0; b < N_BASES; ++b)
        acc += C0[r * N_BASES + b] * B0[(size_t)b * (N_NODES * H_DIM) + (size_t)s * H_DIM + dim];
    atomicAdd(&h0[(size_t)d * H_DIM + dim], nrm * acc);
}

__global__ void relu_f16_kernel(float* __restrict__ h, _Float16* __restrict__ hh, int n) {
    int i = blockIdx.x * blockDim.x + threadIdx.x;
    if (i >= n) return;
    float v = h[i];
    v = v > 0.f ? v : 0.f;
    h[i] = v;
    hh[i] = (_Float16)v;
}

// ---------------------------------------------------------------------------
// WMMA helpers. Fragment layouts used here (wave32, 16x16x32 f16, contiguous K order):
//  A (16x32):  lane<16 row=lane,   halves 0..15 = K 0..15
//              lane>=16 row=lane-16,halves 0..15 = K 16..31
//  B (32x16):  col = lane&15, halves i -> K = (lane>=16 ? 16 : 0) + i
//  C/D (16x16 f32): VGPR j -> M = j + (lane>=16 ? 8 : 0), N = lane&15
// ---------------------------------------------------------------------------
__device__ inline v16h load_A_frag(const _Float16* __restrict__ hrow, int kb, int lane) {
    int khalf = (lane >> 4) * 16;                       // 0 or 16
    v8h lo = *(const v8h*)(hrow + kb + khalf);          // K = kb+khalf .. +7
    v8h hi = *(const v8h*)(hrow + kb + khalf + 8);      // K = kb+khalf+8 .. +15
    return __builtin_shufflevector(lo, hi, 0, 1, 2, 3, 4, 5, 6, 7,
                                           8, 9, 10, 11, 12, 13, 14, 15);
}

__device__ inline v16h load_B_frag(const _Float16* __restrict__ Wt_row, int kb, int lane) {
    // Wt_row points at Wt[...][col][0]; 16 contiguous K halves (32B aligned)
    return *(const v16h*)(Wt_row + kb + ((lane >= 16) ? 16 : 0));
}

// ---------------------------------------------------------------------------
// Dense GEMM: out[n, :OUTD] = hf[n, :64] @ Lt^T + bias.
// One wave handles 4 row-tiles (64 nodes); B fragments register-resident.
// ---------------------------------------------------------------------------
template <int NT>
__global__ __launch_bounds__(32) void dense_wmma_kernel(const _Float16* __restrict__ hf,
                                                        const _Float16* __restrict__ Lt,
                                                        const float* __restrict__ bias,
                                                        float* __restrict__ out) {
    const int OUTD = NT * 16;
    int lane = threadIdx.x;
    int row = lane & 15;

    // weight fragments: shared across all tiles, load once
    v16h bfrag[2][NT];
    float bv[NT];
#pragma unroll
    for (int nt = 0; nt < NT; ++nt) {
        const _Float16* brow = Lt + (size_t)(nt * 16 + row) * H_DIM;
#pragma unroll
        for (int k2 = 0; k2 < 2; ++k2) bfrag[k2][nt] = load_B_frag(brow, k2 * 32, lane);
        bv[nt] = bias[nt * 16 + row];
    }

    int mofs = (lane >= 16) ? 8 : 0;
#pragma unroll
    for (int t = 0; t < 4; ++t) {
        int tIdx = blockIdx.x * 4 + t;
        if (tIdx >= N_NODES / 16) break;               // uniform branch
        int rowBase = tIdx * 16;                       // 50000 = 3125 * 16 exactly
        const _Float16* hrow = hf + (size_t)(rowBase + row) * H_DIM;

        v8f acc[NT];
#pragma unroll
        for (int nt = 0; nt < NT; ++nt) acc[nt] = {};
#pragma unroll
        for (int k2 = 0; k2 < 2; ++k2) {
            v16h a = load_A_frag(hrow, k2 * 32, lane);
#pragma unroll
            for (int nt = 0; nt < NT; ++nt)
                acc[nt] = __builtin_amdgcn_wmma_f32_16x16x32_f16(
                    false, a, false, bfrag[k2][nt], (short)0, acc[nt], false, false);
        }
#pragma unroll
        for (int nt = 0; nt < NT; ++nt) {
            int ncol = nt * 16 + row;
#pragma unroll
            for (int j = 0; j < 8; ++j)
                out[(size_t)(rowBase + j + mofs) * OUTD + ncol] = acc[nt][j] + bv[nt];
        }
    }
}

// ---------------------------------------------------------------------------
// Binned edge transform. One wave handles one 64-edge quad-tile of relation r:
// loads W[r] fragments once into registers, then 4x { gather 16 src rows,
// D = A @ W, norm-scale, atomicAdd scatter to dst }.
// ---------------------------------------------------------------------------
template <int NT>
__global__ __launch_bounds__(32) void edge_wmma_kernel(const _Float16* __restrict__ hf,
                                                       const _Float16* __restrict__ Wt,
                                                       const int* __restrict__ edgeIds,
                                                       const int* __restrict__ src,
                                                       const int* __restrict__ dst,
                                                       const float* __restrict__ norm,
                                                       const int* __restrict__ binOff,
                                                       const int* __restrict__ qtileOff,
                                                       float* __restrict__ out) {
    const int OUTD = NT * 16;
    int qt = blockIdx.x;
    if (qt >= qtileOff[N_RELS]) return;

    // find relation r: largest r with qtileOff[r] <= qt
    int lo = 0, hi = N_RELS - 1;
    while (lo < hi) {
        int mid = (lo + hi + 1) >> 1;
        if (qtileOff[mid] <= qt) lo = mid; else hi = mid - 1;
    }
    int r = lo;
    int eBase = binOff[r] + (qt - qtileOff[r]) * 64;
    int eEnd = binOff[r + 1];

    __shared__ int   s_src[64];
    __shared__ int   s_dst[64];
    __shared__ float s_nrm[64];

    int lane = threadIdx.x;
    int row = lane & 15;

    // stage 64 edges of metadata (2 per lane)
#pragma unroll
    for (int i = lane; i < 64; i += 32) {
        int eIdx = eBase + i;
        if (eIdx < eEnd) {
            int e = edgeIds[eIdx];
            s_src[i] = src[e];
            s_dst[i] = dst[e];
            s_nrm[i] = norm[e];
        } else {                     // padded row: contributes 0
            s_src[i] = 0;
            s_dst[i] = 0;
            s_nrm[i] = 0.f;
        }
    }
    __syncthreads();

    // weight fragments for this relation: register-resident across 4 tiles
    const _Float16* Wr = Wt + (size_t)r * OUTD * H_DIM;
    __builtin_prefetch(Wr, 0, 1);    // global_prefetch_b8 into near caches
    v16h bfrag[2][NT];
#pragma unroll
    for (int nt = 0; nt < NT; ++nt) {
        const _Float16* brow = Wr + (size_t)(nt * 16 + row) * H_DIM;
#pragma unroll
        for (int k2 = 0; k2 < 2; ++k2) bfrag[k2][nt] = load_B_frag(brow, k2 * 32, lane);
    }

    int mofs = (lane >= 16) ? 8 : 0;
#pragma unroll
    for (int t = 0; t < 4; ++t) {
        if (eBase + t * 16 >= eEnd) break;             // uniform branch
        const _Float16* hrow = hf + (size_t)s_src[t * 16 + row] * H_DIM;

        v8f acc[NT];
#pragma unroll
        for (int nt = 0; nt < NT; ++nt) acc[nt] = {};
#pragma unroll
        for (int k2 = 0; k2 < 2; ++k2) {
            v16h a = load_A_frag(hrow, k2 * 32, lane);
#pragma unroll
            for (int nt = 0; nt < NT; ++nt)
                acc[nt] = __builtin_amdgcn_wmma_f32_16x16x32_f16(
                    false, a, false, bfrag[k2][nt], (short)0, acc[nt], false, false);
        }
#pragma unroll
        for (int nt = 0; nt < NT; ++nt) {
            int ncol = nt * 16 + row;
#pragma unroll
            for (int j = 0; j < 8; ++j) {
                int m = t * 16 + j + mofs;
                float v = acc[nt][j] * s_nrm[m];
                atomicAdd(&out[(size_t)s_dst[m] * OUTD + ncol], v);
            }
        }
    }
}

// ---------------------------------------------------------------------------
// Row softmax over 16 classes.
// ---------------------------------------------------------------------------
__global__ void softmax_kernel(const float* __restrict__ inp, float* __restrict__ out) {
    int i = blockIdx.x * blockDim.x + threadIdx.x;
    if (i >= N_NODES) return;
    const float* rp = inp + (size_t)i * OUT_DIM;
    float mx = rp[0];
#pragma unroll
    for (int k = 1; k < OUT_DIM; ++k) mx = fmaxf(mx, rp[k]);
    float ex[OUT_DIM], s = 0.f;
#pragma unroll
    for (int k = 0; k < OUT_DIM; ++k) { ex[k] = __expf(rp[k] - mx); s += ex[k]; }
    float inv = 1.f / s;
#pragma unroll
    for (int k = 0; k < OUT_DIM; ++k) out[(size_t)i * OUT_DIM + k] = ex[k] * inv;
}

// ---------------------------------------------------------------------------
// Host-side orchestration.
// ---------------------------------------------------------------------------
extern "C" void kernel_launch(void* const* d_in, const int* in_sizes, int n_in,
                              void* d_out, int out_size, void* d_ws, size_t ws_size,
                              hipStream_t stream) {
    (void)in_sizes; (void)n_in; (void)out_size; (void)ws_size;

    const int*   src   = (const int*)d_in[1];
    const int*   dst   = (const int*)d_in[2];
    const int*   etype = (const int*)d_in[3];
    const float* norm  = (const float*)d_in[4];
    const float* B0    = (const float*)d_in[5];
    const float* C0    = (const float*)d_in[6];
    const float* L0    = (const float*)d_in[7];
    const float* b0    = (const float*)d_in[8];
    const float* B1    = (const float*)d_in[9];
    const float* C1    = (const float*)d_in[10];
    const float* L1    = (const float*)d_in[11];
    const float* b1    = (const float*)d_in[12];
    const float* B2    = (const float*)d_in[13];
    const float* C2    = (const float*)d_in[14];
    const float* L2    = (const float*)d_in[15];
    const float* b2    = (const float*)d_in[16];
    float* out = (float*)d_out;

    // workspace carve-up (256B aligned)
    char* ws = (char*)d_ws;
    size_t off = 0;
    auto carve = [&](size_t bytes) -> char* {
        char* p = ws + off;
        off = (off + bytes + 255) & ~(size_t)255;
        return p;
    };
    float*    h0       = (float*)carve((size_t)N_NODES * H_DIM * 4);
    float*    h1       = (float*)carve((size_t)N_NODES * H_DIM * 4);
    float*    outp     = (float*)carve((size_t)N_NODES * OUT_DIM * 4);
    _Float16* hh       = (_Float16*)carve((size_t)N_NODES * H_DIM * 2);  // f16 h (reused per layer)
    _Float16* W1t      = (_Float16*)carve((size_t)N_RELS * H_DIM * H_DIM * 2);
    _Float16* W2t      = (_Float16*)carve((size_t)N_RELS * OUT_DIM * H_DIM * 2);
    _Float16* L1t      = (_Float16*)carve((size_t)H_DIM * H_DIM * 2);
    _Float16* L2t      = (_Float16*)carve((size_t)OUT_DIM * H_DIM * 2);
    int*      counts   = (int*)carve(N_RELS * 4);
    int*      cursor   = (int*)carve(N_RELS * 4);
    int*      binOff   = (int*)carve((N_RELS + 1) * 4);
    int*      qtileOff = (int*)carve((N_RELS + 1) * 4);
    int*      edgeIds  = (int*)carve((size_t)N_EDGES * 4);

    const int nQTiles = N_EDGES / 64 + N_RELS;        // upper bound on quad-tiles
    const int denseBlocks = (N_NODES / 16 + 3) / 4;   // 3125 tiles -> 782 waves

    // weights -> f16 transposed
    compute_W_kernel<<<N_RELS, 256, 0, stream>>>(C1, B1, C2, B2, W1t, W2t);
    prep_L_kernel<<<1, 256, 0, stream>>>(L1, L2, L1t, L2t);

    // bin edges by relation
    zero64_kernel<<<1, 64, 0, stream>>>(counts, cursor);
    hist_kernel<<<(N_EDGES + 255) / 256, 256, 0, stream>>>(etype, counts);
    scan_kernel<<<1, 1, 0, stream>>>(counts, binOff, qtileOff);
    scatter_kernel<<<(N_EDGES + 255) / 256, 256, 0, stream>>>(etype, binOff, cursor, edgeIds);

    // ---- layer 0 ----
    init_h0_kernel<<<(N_NODES * H_DIM + 255) / 256, 256, 0, stream>>>(L0, b0, h0);
    edge_layer0_kernel<<<(int)(((long long)N_EDGES * H_DIM + 255) / 256), 256, 0, stream>>>(
        src, dst, etype, norm, B0, C0, h0);
    relu_f16_kernel<<<(N_NODES * H_DIM + 255) / 256, 256, 0, stream>>>(h0, hh, N_NODES * H_DIM);

    // ---- layer 1 ----
    dense_wmma_kernel<4><<<denseBlocks, 32, 0, stream>>>(hh, L1t, b1, h1);
    edge_wmma_kernel<4><<<nQTiles, 32, 0, stream>>>(hh, W1t, edgeIds, src, dst, norm,
                                                    binOff, qtileOff, h1);
    relu_f16_kernel<<<(N_NODES * H_DIM + 255) / 256, 256, 0, stream>>>(h1, hh, N_NODES * H_DIM);

    // ---- layer 2 + softmax ----
    dense_wmma_kernel<1><<<denseBlocks, 32, 0, stream>>>(hh, L2t, b2, outp);
    edge_wmma_kernel<1><<<nQTiles, 32, 0, stream>>>(hh, W2t, edgeIds, src, dst, norm,
                                                    binOff, qtileOff, outp);
    softmax_kernel<<<(N_NODES + 255) / 256, 256, 0, stream>>>(outp, out);
}
